// DimeNet_52922587022003
// MI455X (gfx1250) — matrix operand order repaired and checked by reading, not compiled, not run on hardware
//
#include <hip/hip_runtime.h>
#include <hip/hip_bf16.h>

typedef __attribute__((ext_vector_type(16))) __bf16 v16bf;
typedef __attribute__((ext_vector_type(8)))  float  v8f;

union FragBF {
  v16bf v;
  unsigned int u[8];
  uint4 q[2];
};

__device__ __forceinline__ unsigned short f2bf(float x){
  unsigned int u = __float_as_uint(x);
  u += 0x7fffu + ((u >> 16) & 1u);          // round-to-nearest-even
  return (unsigned short)(u >> 16);
}
__device__ __forceinline__ unsigned int pk2(float a, float b){
  return (unsigned int)f2bf(a) | ((unsigned int)f2bf(b) << 16);
}
__device__ __forceinline__ float swishf(float t){
  return t / (1.0f + __expf(-t));
}
__device__ __forceinline__ v8f wmma_bf16(const FragBF& a, const FragBF& b, v8f c){
  return __builtin_amdgcn_wmma_f32_16x16x32_bf16(false, a.v, false, b.v, (short)0, c, false, false);
}
// A fragment (16x32) built from an fp32 row-major row; K base = kt*32
__device__ __forceinline__ void fragA_f32(const float* row, int kt, int half, FragBF& a){
  const float4* p0 = (const float4*)(row + kt*32 + 16*half);
  const float4* p1 = (const float4*)(row + kt*32 + 8 + 16*half);
  float4 f0 = p0[0], f1 = p0[1], g0 = p1[0], g1 = p1[1];
  a.u[0]=pk2(f0.x,f0.y); a.u[1]=pk2(f0.z,f0.w);
  a.u[2]=pk2(f1.x,f1.y); a.u[3]=pk2(f1.z,f1.w);
  a.u[4]=pk2(g0.x,g0.y); a.u[5]=pk2(g0.z,g0.w);
  a.u[6]=pk2(g1.x,g1.y); a.u[7]=pk2(g1.z,g1.w);
}
// A fragment from padded LDS tile (stride 136 bf16 per row)
__device__ __forceinline__ void fragA_lds(const unsigned short* lds, int mrow, int kt, int half, FragBF& a){
  const unsigned short* r = lds + mrow*136;
  a.q[0] = *(const uint4*)(r + kt*32 + 16*half);
  a.q[1] = *(const uint4*)(r + kt*32 + 8 + 16*half);
}
// B fragment (32x16): p points at W[n_row*Kdim + kt*32 + half*16] (bf16, row-major W[N][K])
__device__ __forceinline__ void fragB_bf16(const unsigned short* p, FragBF& b){
  b.q[0] = *(const uint4*)(p);
  b.q[1] = *(const uint4*)(p + 8);
}

// ---------------- utility kernels ----------------
__global__ void k_fill0(float* __restrict__ p, long n){
  long i = (long)blockIdx.x*blockDim.x + threadIdx.x;
  long s = (long)gridDim.x*blockDim.x;
  for (; i < n; i += s) p[i] = 0.0f;
}
__global__ void k_cvt_bf16(const float* __restrict__ s, unsigned short* __restrict__ d, int n){
  int i = blockIdx.x*blockDim.x + threadIdx.x;
  if (i < n) d[i] = f2bf(s[i]);
}
// pad [rows,6] fp32 -> [rows,32] bf16 (zeros beyond k=5), B-operand friendly
__global__ void k_pad6(const float* __restrict__ s, unsigned short* __restrict__ d, int rows){
  int i = blockIdx.x*blockDim.x + threadIdx.x;
  if (i < rows*32){
    int n = i >> 5, k = i & 31;
    d[i] = (k < 6) ? f2bf(s[n*6 + k]) : (unsigned short)0;
  }
}
// per-edge distance -> Bessel basis with polynomial envelope (p=5)
__global__ void k_edge_prep(const float* __restrict__ pos, const float* __restrict__ freq,
                            const int* __restrict__ esrc, const int* __restrict__ edst,
                            float* __restrict__ rbf, int E){
  int e = blockIdx.x*blockDim.x + threadIdx.x;
  if (e >= E) return;
  int s = esrc[e], t = edst[e];
  float dx = pos[(long)t*3+0] - pos[(long)s*3+0];
  float dy = pos[(long)t*3+1] - pos[(long)s*3+1];
  float dz = pos[(long)t*3+2] - pos[(long)s*3+2];
  float d  = sqrtf(dx*dx + dy*dy + dz*dz);
  float dn = d * 0.2f;                          // d / CUTOFF
  float d2 = dn*dn;
  float dp = d2*d2*dn;                          // dn^5
  float env = 1.0f/dn + dp*(-21.0f + dn*(35.0f - 15.0f*dn));
  #pragma unroll
  for (int r = 0; r < 6; ++r)
    rbf[(long)e*6 + r] = env * sinf(freq[r]*dn);
}
// triplet angles
__global__ void k_angle(const float* __restrict__ pos, const int* __restrict__ ii,
                        const int* __restrict__ jj, const int* __restrict__ kk,
                        float* __restrict__ ang, int T){
  int t = blockIdx.x*blockDim.x + threadIdx.x;
  if (t >= T) return;
  int i = ii[t], j = jj[t], k = kk[t];
  float pix = pos[(long)i*3+0], piy = pos[(long)i*3+1], piz = pos[(long)i*3+2];
  float ax = pos[(long)j*3+0]-pix, ay = pos[(long)j*3+1]-piy, az = pos[(long)j*3+2]-piz;
  float bx = pos[(long)k*3+0]-pix, by = pos[(long)k*3+1]-piy, bz = pos[(long)k*3+2]-piz;
  float dot = ax*bx + ay*by + az*bz;
  float cx = ay*bz - az*by, cy = az*bx - ax*bz, cz = ax*by - ay*bx;
  float cr = sqrtf(cx*cx + cy*cy + cz*cz);
  ang[t] = atan2f(cr, dot);
}

// ---------------- h = x @ Wx^T  -> bf16 ----------------
__global__ __launch_bounds__(32) void k_embed(const float* __restrict__ x,
                                              const unsigned short* __restrict__ Wxbf,
                                              unsigned short* __restrict__ hbf, int N){
  const int lane = threadIdx.x, mrow = lane & 15, half = lane >> 4, col = mrow;
  const int n0 = blockIdx.x * 16;
  int row = n0 + mrow; if (row >= N) row = N - 1;
  const float* xrow = x + (long)row*128;
  v8f acc[8];
  #pragma unroll
  for (int nt = 0; nt < 8; ++nt)
    #pragma unroll
    for (int i = 0; i < 8; ++i) acc[nt][i] = 0.0f;
  #pragma unroll
  for (int kt = 0; kt < 4; ++kt){
    FragBF A; fragA_f32(xrow, kt, half, A);
    #pragma unroll
    for (int nt = 0; nt < 8; ++nt){
      FragBF B; fragB_bf16(Wxbf + (nt*16 + col)*128 + kt*32 + half*16, B);
      acc[nt] = wmma_bf16(A, B, acc[nt]);
    }
  }
  #pragma unroll
  for (int nt = 0; nt < 8; ++nt)
    #pragma unroll
    for (int v = 0; v < 8; ++v){
      int m = n0 + v + 8*half;
      if (m < N) hbf[(long)m*128 + nt*16 + col] = f2bf(acc[nt][v]);
    }
}

// ------- fused edge pipeline (8 waves / block, 1 n-tile per wave, register-resident Wl) -------
#define EDGE_TILES_PER_BLOCK 16
__global__ __launch_bounds__(256) void k_edge_fused(
    const unsigned short* __restrict__ hbf,  const unsigned short* __restrict__ Wlbf,
    const unsigned short* __restrict__ WrB,  const unsigned short* __restrict__ WrbfB,
    const float* __restrict__ rbf, const float* __restrict__ br, const float* __restrict__ bl,
    const int* __restrict__ esrc,  const int* __restrict__ edst,
    float* __restrict__ node, int E){
  __shared__ unsigned short ldsD[16*136];   // h[dst] tile (bf16)
  __shared__ unsigned short ldsS[16*136];   // h[src] tile (bf16)
  __shared__ unsigned short ldsR[16*136];   // rbf_h tile (bf16)
  const int tid  = threadIdx.x;
  const int lane = tid & 31;
  const int nt   = tid >> 5;                // wave id == n-tile (0..7)
  const int mrow = lane & 15, half = lane >> 4, col = mrow;
  const int ncol = nt*16 + col;

  // persistent register-resident B fragments (Wl slice: 12 k-steps, plus Wr / Wrbf)
  FragBF Bwl[12];
  #pragma unroll
  for (int kt = 0; kt < 12; ++kt)
    fragB_bf16(Wlbf + ncol*384 + kt*32 + half*16, Bwl[kt]);
  FragBF Bwr;   fragB_bf16(WrB   + ncol*32 + half*16, Bwr);
  FragBF Bwrbf; fragB_bf16(WrbfB + ncol*32 + half*16, Bwrbf);
  const float brv = br[ncol];
  const float blv = bl[ncol];

  const long base = (long)blockIdx.x * (16*EDGE_TILES_PER_BLOCK);

  for (int tile = 0; tile < EDGE_TILES_PER_BLOCK; ++tile){
    const long e0 = base + (long)tile*16;
    __syncthreads();                        // previous tile's LDS reads done
    // ---- cooperative staging: h[dst]/h[src] 16x128 bf16 tiles, one b128 per thread ----
    {
      int c   = tid & 127;                  // chunk 0..127
      int row = c >> 3, seg = c & 7;
      long ee = e0 + row; if (ee >= E) ee = E - 1;
      int nid = (tid < 128) ? edst[ee] : esrc[ee];
      const uint4* g = (const uint4*)(hbf + (long)nid*128 + seg*16);
      uint4* l = (uint4*)((tid < 128 ? ldsD : ldsS) + row*136 + seg*16);
      *l = *g;
    }
    // ---- per-wave A fragment of rbf (K=0..5 live, zero-padded to 32) ----
    long e = e0 + mrow; if (e >= E) e = E - 1;
    FragBF Arbf;
    #pragma unroll
    for (int i = 0; i < 8; ++i) Arbf.u[i] = 0u;
    if (half == 0){
      float r0 = rbf[e*6+0], r1 = rbf[e*6+1], r2 = rbf[e*6+2];
      float r3 = rbf[e*6+3], r4 = rbf[e*6+4], r5 = rbf[e*6+5];
      Arbf.u[0] = pk2(r0, r1);
      Arbf.u[1] = pk2(r2, r3);
      Arbf.u[2] = pk2(r4, r5);
    }
    // ---- phase 1: this wave's 16-col slice of rbf_h = swish(rbf @ Wr^T + br) ----
    v8f c1;
    #pragma unroll
    for (int i = 0; i < 8; ++i) c1[i] = brv;
    v8f dh = wmma_bf16(Arbf, Bwr, c1);
    #pragma unroll
    for (int v = 0; v < 8; ++v)
      ldsR[(v + 8*half)*136 + ncol] = f2bf(swishf(dh[v]));
    __syncthreads();                        // staging + rbf_h visible to all waves
    // ---- phase 2: acc = [h_dst | h_src | rbf_h] @ Wl^T + bl  (K=384) ----
    v8f acc;
    #pragma unroll
    for (int i = 0; i < 8; ++i) acc[i] = blv;
    #pragma unroll
    for (int kt = 0; kt < 4; ++kt){
      FragBF A; fragA_lds(ldsD, mrow, kt, half, A);
      acc = wmma_bf16(A, Bwl[kt], acc);
    }
    #pragma unroll
    for (int kt = 0; kt < 4; ++kt){
      FragBF A; fragA_lds(ldsS, mrow, kt, half, A);
      acc = wmma_bf16(A, Bwl[4+kt], acc);
    }
    #pragma unroll
    for (int kt = 0; kt < 4; ++kt){
      FragBF A; fragA_lds(ldsR, mrow, kt, half, A);
      acc = wmma_bf16(A, Bwl[8+kt], acc);
    }
    // ---- phase 3: t = (rbf @ Wrbf^T) * swish(acc) ; node[dst] += t ----
    v8f z;
    #pragma unroll
    for (int i = 0; i < 8; ++i) z[i] = 0.0f;
    v8f d2 = wmma_bf16(Arbf, Bwrbf, z);
    #pragma unroll
    for (int v = 0; v < 8; ++v){
      long ee = e0 + 8*half + v;
      if (ee < E){
        int did = edst[ee];
        float tv = d2[v] * swishf(acc[v]);
        unsafeAtomicAdd(node + (long)did*128 + nt*16 + col, tv);
      }
    }
  }
}

// ---------------- node MLP (3x swish layers) + Wout ----------------
__global__ __launch_bounds__(32) void k_node_mlp(const float* __restrict__ node,
                                                 const unsigned short* __restrict__ Wlsbf,
                                                 const float* __restrict__ bls,
                                                 const unsigned short* __restrict__ Woutbf,
                                                 float* __restrict__ P, int N){
  __shared__ unsigned short lds[16*136];
  const int lane = threadIdx.x, mrow = lane & 15, half = lane >> 4, col = mrow;
  const int n0 = blockIdx.x * 16;
  int row = n0 + mrow; if (row >= N) row = N - 1;
  const float* nrow = node + (long)row*128;

  v8f acc[8];
  // layer 0: A from fp32 node accumulator
  #pragma unroll
  for (int nt = 0; nt < 8; ++nt){
    float bv = bls[nt*16 + col];
    #pragma unroll
    for (int i = 0; i < 8; ++i) acc[nt][i] = bv;
  }
  #pragma unroll
  for (int kt = 0; kt < 4; ++kt){
    FragBF A; fragA_f32(nrow, kt, half, A);
    #pragma unroll
    for (int nt = 0; nt < 8; ++nt){
      FragBF B; fragB_bf16(Wlsbf + (nt*16 + col)*128 + kt*32 + half*16, B);
      acc[nt] = wmma_bf16(A, B, acc[nt]);
    }
  }
  #pragma unroll
  for (int nt = 0; nt < 8; ++nt)
    #pragma unroll
    for (int v = 0; v < 8; ++v)
      lds[(v + 8*half)*136 + nt*16 + col] = f2bf(swishf(acc[nt][v]));
  __syncthreads();

  for (int L = 1; L < 3; ++L){
    const unsigned short* WL = Wlsbf + L*128*128;
    #pragma unroll
    for (int nt = 0; nt < 8; ++nt){
      float bv = bls[L*128 + nt*16 + col];
      #pragma unroll
      for (int i = 0; i < 8; ++i) acc[nt][i] = bv;
    }
    #pragma unroll
    for (int kt = 0; kt < 4; ++kt){
      FragBF A; fragA_lds(lds, mrow, kt, half, A);
      #pragma unroll
      for (int nt = 0; nt < 8; ++nt){
        FragBF B; fragB_bf16(WL + (nt*16 + col)*128 + kt*32 + half*16, B);
        acc[nt] = wmma_bf16(A, B, acc[nt]);
      }
    }
    __syncthreads();   // all reads of the tile done before overwriting
    #pragma unroll
    for (int nt = 0; nt < 8; ++nt)
      #pragma unroll
      for (int v = 0; v < 8; ++v)
        lds[(v + 8*half)*136 + nt*16 + col] = f2bf(swishf(acc[nt][v]));
    __syncthreads();
  }

  // output: P = L3 @ Wout^T   (OUT = 64 -> 4 N-tiles)
  v8f o[4];
  #pragma unroll
  for (int nt = 0; nt < 4; ++nt)
    #pragma unroll
    for (int i = 0; i < 8; ++i) o[nt][i] = 0.0f;
  #pragma unroll
  for (int kt = 0; kt < 4; ++kt){
    FragBF A; fragA_lds(lds, mrow, kt, half, A);
    #pragma unroll
    for (int nt = 0; nt < 4; ++nt){
      FragBF B; fragB_bf16(Woutbf + (nt*16 + col)*128 + kt*32 + half*16, B);
      o[nt] = wmma_bf16(A, B, o[nt]);
    }
  }
  #pragma unroll
  for (int nt = 0; nt < 4; ++nt)
    #pragma unroll
    for (int v = 0; v < 8; ++v){
      int m = n0 + v + 8*half;
      if (m < N) P[(long)m*64 + nt*16 + col] = o[nt][v];
    }
}

extern "C" void kernel_launch(void* const* d_in, const int* in_sizes, int n_in,
                              void* d_out, int out_size, void* d_ws, size_t ws_size,
                              hipStream_t stream){
  (void)n_in; (void)out_size; (void)ws_size;
  const float* x    = (const float*)d_in[0];
  const float* pos  = (const float*)d_in[1];
  const float* freq = (const float*)d_in[2];
  const float* Wx   = (const float*)d_in[3];
  const float* Wr   = (const float*)d_in[4];
  const float* br   = (const float*)d_in[5];
  const float* Wl   = (const float*)d_in[6];
  const float* bl   = (const float*)d_in[7];
  const float* Wrbf = (const float*)d_in[8];
  const float* Wls  = (const float*)d_in[9];
  const float* bls  = (const float*)d_in[10];
  const float* Wout = (const float*)d_in[11];
  const int* esrc = (const int*)d_in[12];
  const int* edst = (const int*)d_in[13];
  const int* ii   = (const int*)d_in[14];
  const int* jj   = (const int*)d_in[15];
  const int* kk   = (const int*)d_in[16];
  const int N = in_sizes[0] / 128;
  const int E = in_sizes[12];
  const int T = in_sizes[14];

  char* w = (char*)d_ws;
  auto take = [&](size_t bytes) -> char* {
    char* p = w;
    w += (bytes + 255) & ~(size_t)255;
    return p;
  };
  unsigned short* hbf    = (unsigned short*)take((size_t)N*128*2);
  unsigned short* Wxbf   = (unsigned short*)take((size_t)128*128*2);
  unsigned short* Wlbf   = (unsigned short*)take((size_t)128*384*2);
  unsigned short* Wlsbf  = (unsigned short*)take((size_t)3*128*128*2);
  unsigned short* Woutbf = (unsigned short*)take((size_t)64*128*2);
  unsigned short* WrB    = (unsigned short*)take((size_t)128*32*2);
  unsigned short* WrbfB  = (unsigned short*)take((size_t)128*32*2);
  float* rbfws  = (float*)take((size_t)E*6*4);
  float* nodews = (float*)take((size_t)N*128*4);

  float* P   = (float*)d_out;
  float* ang = P + (size_t)N*64;

  k_cvt_bf16<<<(128*128 + 255)/256, 256, 0, stream>>>(Wx,  Wxbf,  128*128);
  k_cvt_bf16<<<(128*384 + 255)/256, 256, 0, stream>>>(Wl,  Wlbf,  128*384);
  k_cvt_bf16<<<(3*128*128 + 255)/256, 256, 0, stream>>>(Wls, Wlsbf, 3*128*128);
  k_cvt_bf16<<<(64*128 + 255)/256, 256, 0, stream>>>(Wout, Woutbf, 64*128);
  k_pad6<<<(128*32 + 255)/256, 256, 0, stream>>>(Wr,   WrB,   128);
  k_pad6<<<(128*32 + 255)/256, 256, 0, stream>>>(Wrbf, WrbfB, 128);
  k_fill0<<<1024, 256, 0, stream>>>(nodews, (long)N*128);
  k_edge_prep<<<(E + 255)/256, 256, 0, stream>>>(pos, freq, esrc, edst, rbfws, E);
  k_embed<<<(N + 15)/16, 32, 0, stream>>>(x, Wxbf, hbf, N);
  {
    int edges_per_block = 16*EDGE_TILES_PER_BLOCK;
    int nblk = (E + edges_per_block - 1) / edges_per_block;
    k_edge_fused<<<nblk, 256, 0, stream>>>(hbf, Wlbf, WrB, WrbfB, rbfws,
                                           br, bl, esrc, edst, nodews, E);
  }
  k_node_mlp<<<(N + 15)/16, 32, 0, stream>>>(nodews, Wlsbf, bls, Woutbf, P, N);
  k_angle<<<(T + 255)/256, 256, 0, stream>>>(pos, ii, jj, kk, ang, T);
}
